// SelfAttention_72773925863461
// MI455X (gfx1250) — compile-verified, matrix operand order checked by reading, not run on hardware
//
#include <hip/hip_runtime.h>
#include <hip/hip_bf16.h>

typedef __bf16 bf16;
typedef __attribute__((ext_vector_type(16))) __bf16 bf16x16;
typedef __attribute__((ext_vector_type(8)))  __bf16 bf16x8;
typedef __attribute__((ext_vector_type(4)))  __bf16 bf16x4;
typedef __attribute__((ext_vector_type(8)))  float   f32x8;
typedef __attribute__((ext_vector_type(4)))  float   f32x4;

#define BATCH 8
#define CCH   1024
#define NTOK  4096
#define RCH   128
#define SCALE 0.08838834764831845f   // 1/sqrt(128)

#if defined(__has_builtin)
#if __has_builtin(__builtin_amdgcn_tensor_load_to_lds) && __has_builtin(__builtin_amdgcn_s_wait_tensorcnt)
#define USE_TDM 1
#endif
#endif
#ifndef USE_TDM
#define USE_TDM 0
#endif

// ---------------------------------------------------------------------------
// WMMA helper: D = A(16x32 bf16) * B(32x16 bf16) + C(16x16 f32)
// ---------------------------------------------------------------------------
__device__ __forceinline__ f32x8 wmma_bf16(bf16x16 a, bf16x16 b, f32x8 c) {
  return __builtin_amdgcn_wmma_f32_16x16x32_bf16(
      false, a, false, b, (short)0, c, false, false);
}

// 16x32 bf16 A-fragment (or B^T-fragment) from row-major [16][stride] tile.
// Two 16-byte loads per lane (ds_load_b128 / global_load_b128).
__device__ __forceinline__ bf16x16 load_frag(const bf16* __restrict__ p0, int stride) {
  const int lane = threadIdx.x & 31;
  const bf16* p = p0 + (lane & 15) * stride + (lane >> 4) * 8;
  union { bf16x16 v; bf16x8 h[2]; } u;
  u.h[0] = *(const bf16x8*)(p);
  u.h[1] = *(const bf16x8*)(p + 16);
  return u.v;
}

// ---------------------------------------------------------------------------
// Tensor Data Mover: async 2D bf16 tile -> LDS (with optional row padding).
// One wave issues; caller does s_wait_tensorcnt + workgroup barrier.
// ---------------------------------------------------------------------------
__device__ __forceinline__ unsigned lds_off(const void* p) {
  // generic LDS pointer: low 32 bits are the LDS byte offset (ISA 10.2 aperture)
  return (unsigned)(size_t)p;
}

#if USE_TDM
typedef __attribute__((ext_vector_type(4))) unsigned int u32x4;
typedef __attribute__((ext_vector_type(8))) int          i32x8;
typedef __attribute__((ext_vector_type(4))) int          i32x4;

__device__ __forceinline__ void tdm_load_2d(unsigned ldsOff, const void* gptr,
                                            unsigned tileD0, unsigned tileD1,
                                            unsigned long long strideD0,
                                            unsigned tensorD0, unsigned tensorD1,
                                            unsigned padIntEnc, unsigned padAmtEnc)
{
  unsigned long long ga = (unsigned long long)gptr;
  u32x4 g0;
  g0[0] = 1u;                                               // count=1, user D#
  g0[1] = ldsOff;                                           // lds_addr (bytes)
  g0[2] = (unsigned)ga;                                     // global_addr lo
  g0[3] = ((unsigned)(ga >> 32) & 0x1FFFFFFu) | (2u << 30); // addr hi | type=2
  i32x8 g1;
  g1[0] = (int)((1u << 16)                 // data_size = 2 bytes
              | (1u << 20)                 // pad_enable
              | (padIntEnc << 22) | (padAmtEnc << 25));
  g1[1] = (int)((tensorD0 & 0xFFFFu) << 16);                        // dim0 lo16
  g1[2] = (int)(((tensorD0 >> 16) & 0xFFFFu) | ((tensorD1 & 0xFFFFu) << 16));
  g1[3] = (int)(((tensorD1 >> 16) & 0xFFFFu) | ((tileD0 & 0xFFFFu) << 16));
  g1[4] = (int)(tileD1 & 0xFFFFu);                                  // tile_dim1
  g1[5] = (int)(unsigned)(strideD0 & 0xFFFFFFFFull);                // stride lo
  g1[6] = (int)(unsigned)((strideD0 >> 32) & 0xFFFFu);              // stride hi
  g1[7] = 0;
  i32x4 gz = {0, 0, 0, 0};
#if __clang_major__ >= 23
  i32x8 gz8 = {0, 0, 0, 0, 0, 0, 0, 0};
  __builtin_amdgcn_tensor_load_to_lds(g0, g1, gz, gz, gz8, 0);
#else
  __builtin_amdgcn_tensor_load_to_lds(g0, g1, gz, gz, 0);
#endif
}
#endif

// ---------------------------------------------------------------------------
// Pre-pass 1: x (f32, [b][c][n]) -> xT (bf16, [b][n][c]) via LDS transpose.
// ---------------------------------------------------------------------------
__global__ __launch_bounds__(256)
void transpose_to_bf16(const float* __restrict__ x, bf16* __restrict__ xT)
{
  __shared__ bf16 T[64 * 72];                 // [n][c] tile
  const int tid = threadIdx.x;
  const int b  = blockIdx.z;
  const int c0 = blockIdx.y * 64;
  const int n0 = blockIdx.x * 64;

  f32x4 vv[4];
#pragma unroll
  for (int k = 0; k < 4; ++k) {               // batch all loads first
    int idx = k * 256 + tid;                  // 1024 float4 chunks
    int cc = idx >> 4, nq = (idx & 15) * 4;
    vv[k] = *(const f32x4*)&x[((size_t)b * CCH + c0 + cc) * NTOK + n0 + nq];
  }
#pragma unroll
  for (int k = 0; k < 4; ++k) {               // then scatter transposed
    int idx = k * 256 + tid;
    int cc = idx >> 4, nq = (idx & 15) * 4;
#pragma unroll
    for (int u = 0; u < 4; ++u) T[(nq + u) * 72 + cc] = (bf16)vv[k][u];
  }
  __syncthreads();
#pragma unroll
  for (int k = 0; k < 2; ++k) {               // coalesced b128 writes
    int idx = k * 256 + tid;                  // 512 bf16x8 chunks
    int nn = idx >> 3, cq = (idx & 7) * 8;
    bf16x8 pk = *(const bf16x8*)&T[nn * 72 + cq];
    *(bf16x8*)&xT[((size_t)b * NTOK + n0 + nn) * CCH + c0 + cq] = pk;
  }
}

// ---------------------------------------------------------------------------
// Pre-pass 2: elementwise f32 -> bf16 (weights).
// ---------------------------------------------------------------------------
__global__ __launch_bounds__(256)
void cvt_f32_bf16(const float* __restrict__ in, bf16* __restrict__ o, int nelem)
{
  int i = (blockIdx.x * 256 + threadIdx.x) * 4;
  if (i < nelem) {
    f32x4 v = *(const f32x4*)&in[i];
    bf16x4 pk;
#pragma unroll
    for (int u = 0; u < 4; ++u) pk[u] = (bf16)v[u];
    *(bf16x4*)&o[i] = pk;
  }
}

// ---------------------------------------------------------------------------
// Projection GEMM:  out[token, j] = sum_c xT[b,n,c] * Wb[j,c] + bias[j]
// Tiles staged to LDS by TDM (padded rows), 1 WMMA per tile per k-step.
// chanMajor=0 -> bf16 token-major [b][n][j] (Q,K);  1 -> channel-major (V^T)
// ---------------------------------------------------------------------------
__global__ __launch_bounds__(256)
void proj_bf16_gemm(const bf16* __restrict__ xT, const bf16* __restrict__ Wb,
                    const float* __restrict__ bias, bf16* __restrict__ out,
                    int Cin, int Nout, int chanMajor)
{
  __shared__ bf16 As[64 * 40];    // [token][k], 32+8 pad
  __shared__ bf16 Bs[128 * 40];   // [j][k]

  const int tid  = threadIdx.x;
  const int lane = tid & 31;
  const int half = lane >> 4;
  const int wave = tid >> 5;
  const int it   = wave & 3;
  const int jh   = wave >> 2;
  const int b    = blockIdx.z;
  const int n0   = blockIdx.x * 64;
  const int j0   = blockIdx.y * 128;

  f32x8 z = {0.f,0.f,0.f,0.f,0.f,0.f,0.f,0.f};
  f32x8 acc[4] = {z, z, z, z};

  for (int c0 = 0; c0 < Cin; c0 += 32) {
#if USE_TDM
    if (wave == 0) {
      // 64x32 and 128x32 bf16 tiles; rows padded 64B->80B (int=16dw, amt=4dw)
      tdm_load_2d(lds_off(As), xT + ((size_t)b * NTOK + n0) * Cin + c0,
                  32, 64, (unsigned long long)Cin, (unsigned)Cin, NTOK, 3, 3);
      tdm_load_2d(lds_off(Bs), Wb + (size_t)j0 * Cin + c0,
                  32, 128, (unsigned long long)Cin, (unsigned)Cin, Nout, 3, 3);
      __builtin_amdgcn_s_wait_tensorcnt((short)0);
    }
#else
    {   // fallback: phase-split b128 copies
      int n = tid >> 2, cq = (tid & 3) * 8;   // 256 chunks of 8
      bf16x8 va = *(const bf16x8*)&xT[((size_t)b * NTOK + n0 + n) * Cin + c0 + cq];
      bf16x8 vb0, vb1;
      {
        int idx = tid;          int jj = idx >> 2, cc = (idx & 3) * 8;
        vb0 = *(const bf16x8*)&Wb[(size_t)(j0 + jj) * Cin + c0 + cc];
        int idx2 = 256 + tid;   int jj2 = idx2 >> 2, cc2 = (idx2 & 3) * 8;
        vb1 = *(const bf16x8*)&Wb[(size_t)(j0 + jj2) * Cin + c0 + cc2];
        *(bf16x8*)&As[n * 40 + cq] = va;
        *(bf16x8*)&Bs[jj * 40 + cc] = vb0;
        *(bf16x8*)&Bs[jj2 * 40 + cc2] = vb1;
      }
    }
#endif
    __syncthreads();

    bf16x16 a = load_frag(As + it * 16 * 40, 40);
#pragma unroll
    for (int t = 0; t < 4; ++t) {
      bf16x16 bb = load_frag(Bs + (jh * 64 + t * 16) * 40, 40);
      acc[t] = wmma_bf16(a, bb, acc[t]);
    }
    __syncthreads();
  }

  // epilogue: C-fragment row i = v + 8*half, col = lane&15
#pragma unroll
  for (int t = 0; t < 4; ++t) {
    const int jg = j0 + jh * 64 + t * 16 + (lane & 15);
    const float bv = bias[jg];
    if (chanMajor) {
      size_t base = ((size_t)b * Nout + jg) * NTOK + n0 + it * 16 + half * 8;
      bf16x8 pk;
#pragma unroll
      for (int v = 0; v < 8; ++v) pk[v] = (bf16)(acc[t][v] + bv);
      *(bf16x8*)&out[base] = pk;
    } else {
#pragma unroll
      for (int v = 0; v < 8; ++v) {
        int n = n0 + it * 16 + half * 8 + v;
        out[((size_t)b * NTOK + n) * Nout + jg] = (bf16)(acc[t][v] + bv);
      }
    }
  }
}

// ---------------------------------------------------------------------------
// Flash attention: 32 queries x full C per block, 8 waves.
// K tiles arrive via TDM each iteration; V read channel-major from global.
// ---------------------------------------------------------------------------
__global__ __launch_bounds__(256)
void flash_attn(const bf16* __restrict__ Q, const bf16* __restrict__ K,
                const bf16* __restrict__ Vt, const float* __restrict__ x,
                const float* __restrict__ gamma, float* __restrict__ out)
{
  __shared__ bf16  Qs[32 * 136];
  __shared__ bf16  Ks[64 * 136];
  __shared__ float Ss[32 * 68];
  __shared__ bf16  Ps[32 * 72];
  __shared__ float mrow[32], lrow[32], arow[32];

  const int tid  = threadIdx.x;
  const int lane = tid & 31;
  const int half = lane >> 4;
  const int wave = tid >> 5;
  const int b    = blockIdx.y;
  const int i0   = blockIdx.x * 32;
  const int s_it = wave & 1;
  const int s_jt = wave >> 1;
  const int cb   = wave * 128;

  // stage Q once: 32x128 tile, padded rows 256B->272B (int=64dw, amt=4dw)
#if USE_TDM
  if (wave == 0) {
    tdm_load_2d(lds_off(Qs), Q + ((size_t)b * NTOK + i0) * RCH,
                RCH, 32, RCH, RCH, NTOK, 5, 3);
    __builtin_amdgcn_s_wait_tensorcnt((short)0);
  }
#else
#pragma unroll
  for (int k = 0; k < 2; ++k) {
    int idx = k * 256 + tid;                   // 512 bf16x8 chunks
    int ii = idx >> 4, cq = (idx & 15) * 8;
    *(bf16x8*)&Qs[ii * 136 + cq] =
        *(const bf16x8*)&Q[((size_t)b * NTOK + i0 + ii) * RCH + cq];
  }
#endif
  if (tid < 32) { mrow[tid] = -1e30f; lrow[tid] = 0.f; }

  f32x8 z = {0.f,0.f,0.f,0.f,0.f,0.f,0.f,0.f};
  f32x8 acc[16];
#pragma unroll
  for (int f = 0; f < 16; ++f) acc[f] = z;
  __syncthreads();

  for (int j0 = 0; j0 < NTOK; j0 += 64) {
#if USE_TDM
    if (wave == 0) {
      tdm_load_2d(lds_off(Ks), K + ((size_t)b * NTOK + j0) * RCH,
                  RCH, 64, RCH, RCH, NTOK, 5, 3);
      __builtin_amdgcn_s_wait_tensorcnt((short)0);
    }
#else
#pragma unroll
    for (int k = 0; k < 4; ++k) {
      int idx = k * 256 + tid;                 // 1024 bf16x8 chunks
      int jj = idx >> 4, cq = (idx & 15) * 8;
      *(bf16x8*)&Ks[jj * 136 + cq] =
          *(const bf16x8*)&K[((size_t)b * NTOK + j0 + jj) * RCH + cq];
    }
#endif
    __syncthreads();

    // S tile = scale * Q K^T  (R=128 -> 4 WMMA k-steps per wave)
    f32x8 s = z;
#pragma unroll
    for (int kk = 0; kk < 4; ++kk) {
      bf16x16 a  = load_frag(Qs + s_it * 16 * 136 + kk * 32, 136);
      bf16x16 bb = load_frag(Ks + s_jt * 16 * 136 + kk * 32, 136);
      s = wmma_bf16(a, bb, s);
    }
    {
      const int col = s_jt * 16 + (lane & 15);
#pragma unroll
      for (int v = 0; v < 8; ++v)
        Ss[(s_it * 16 + v + 8 * half) * 68 + col] = s[v] * SCALE;
    }
    __syncthreads();

    // online softmax: 8 lanes per row, shfl_xor reductions
    {
      const int r  = tid >> 3;
      const int sl = tid & 7;
      float v[8];
      float bm = -1e30f;
#pragma unroll
      for (int k = 0; k < 8; ++k) {
        v[k] = Ss[r * 68 + sl + k * 8];
        bm = fmaxf(bm, v[k]);
      }
#pragma unroll
      for (int m = 1; m < 8; m <<= 1) bm = fmaxf(bm, __shfl_xor(bm, m, 8));
      float mold = mrow[r];
      float mnew = fmaxf(mold, bm);
      float sum = 0.f;
#pragma unroll
      for (int k = 0; k < 8; ++k) {
        float p = __expf(v[k] - mnew);
        sum += p;
        Ps[r * 72 + sl + k * 8] = (bf16)p;
      }
#pragma unroll
      for (int m = 1; m < 8; m <<= 1) sum += __shfl_xor(sum, m, 8);
      if (sl == 0) {
        float al = __expf(mold - mnew);
        lrow[r] = lrow[r] * al + sum;
        mrow[r] = mnew;
        arow[r] = al;
      }
    }
    __syncthreads();

    // rescale accumulators by alpha
    f32x8 al0, al1;
#pragma unroll
    for (int v = 0; v < 8; ++v) {
      al0[v] = arow[v + 8 * half];
      al1[v] = arow[16 + v + 8 * half];
    }
#pragma unroll
    for (int ct = 0; ct < 8; ++ct) {
      acc[ct]     = acc[ct] * al0;
      acc[8 + ct] = acc[8 + ct] * al1;
    }

    // O += P @ V  (2 k-steps x 8 channel tiles x 2 query tiles = 32 WMMA)
#pragma unroll
    for (int kk = 0; kk < 2; ++kk) {
      bf16x16 a0 = load_frag(Ps + kk * 32, 72);
      bf16x16 a1 = load_frag(Ps + 16 * 72 + kk * 32, 72);
#pragma unroll
      for (int ct = 0; ct < 8; ++ct) {
        const bf16* vb = Vt + ((size_t)b * CCH + cb + ct * 16) * NTOK + j0 + kk * 32;
        bf16x16 bb = load_frag(vb, NTOK);
        acc[ct]     = wmma_bf16(a0, bb, acc[ct]);
        acc[8 + ct] = wmma_bf16(a1, bb, acc[8 + ct]);
      }
    }
    __syncthreads();
  }

  // epilogue: out = gamma * (O / l) + x
  f32x8 li0, li1;
#pragma unroll
  for (int v = 0; v < 8; ++v) {
    li0[v] = 1.0f / lrow[v + 8 * half];
    li1[v] = 1.0f / lrow[16 + v + 8 * half];
  }
  const float g = gamma[0];
#pragma unroll
  for (int it = 0; it < 2; ++it) {
#pragma unroll
    for (int ct = 0; ct < 8; ++ct) {
      const int c = cb + ct * 16 + (lane & 15);
      const size_t base = ((size_t)b * CCH + c) * NTOK + i0 + it * 16 + 8 * half;
      f32x8 o = acc[it * 8 + ct] * (it ? li1 : li0);
      const f32x4* xp = (const f32x4*)&x[base];
      f32x4 x0 = xp[0], x1 = xp[1];
      f32x4 r0, r1;
#pragma unroll
      for (int v = 0; v < 4; ++v) {
        r0[v] = g * o[v]     + x0[v];
        r1[v] = g * o[v + 4] + x1[v];
      }
      f32x4* op = (f32x4*)&out[base];
      op[0] = r0;
      op[1] = r1;
    }
  }
}

// ---------------------------------------------------------------------------
extern "C" void kernel_launch(void* const* d_in, const int* in_sizes, int n_in,
                              void* d_out, int out_size, void* d_ws, size_t ws_size,
                              hipStream_t stream) {
  (void)in_sizes; (void)n_in; (void)out_size; (void)ws_size;
  const float* x     = (const float*)d_in[0];
  const float* Wq    = (const float*)d_in[1];
  const float* bq    = (const float*)d_in[2];
  const float* Wk    = (const float*)d_in[3];
  const float* bk    = (const float*)d_in[4];
  const float* Wv    = (const float*)d_in[5];
  const float* bv    = (const float*)d_in[6];
  const float* gamma = (const float*)d_in[7];
  float* out = (float*)d_out;

  const size_t MB = 1024 * 1024;
  bf16* Qw  = (bf16*)d_ws;                               //  8 MB
  bf16* Kw  = (bf16*)((char*)d_ws + 8   * MB);           //  8 MB
  bf16* Vw  = (bf16*)((char*)d_ws + 16  * MB);           // 64 MB (channel-major)
  bf16* xT  = (bf16*)((char*)d_ws + 80  * MB);           // 64 MB ([b][n][c])
  bf16* Wqb = (bf16*)((char*)d_ws + 144 * MB);           // 256 KB
  bf16* Wkb = (bf16*)((char*)d_ws + 144 * MB + 256 * 1024);
  bf16* Wvb = (bf16*)((char*)d_ws + 144 * MB + 512 * 1024); // 2 MB

  dim3 blk(256);
  // pre-passes: transpose/convert activations + weights to bf16
  transpose_to_bf16<<<dim3(NTOK / 64, CCH / 64, BATCH), blk, 0, stream>>>(x, xT);
  cvt_f32_bf16<<<dim3((RCH * CCH) / 1024), blk, 0, stream>>>(Wq, Wqb, RCH * CCH);
  cvt_f32_bf16<<<dim3((RCH * CCH) / 1024), blk, 0, stream>>>(Wk, Wkb, RCH * CCH);
  cvt_f32_bf16<<<dim3((CCH * CCH) / 1024), blk, 0, stream>>>(Wv, Wvb, CCH * CCH);
  // projections (WMMA bf16)
  proj_bf16_gemm<<<dim3(NTOK / 64, 1, BATCH), blk, 0, stream>>>(xT, Wqb, bq, Qw, CCH, RCH, 0);
  proj_bf16_gemm<<<dim3(NTOK / 64, 1, BATCH), blk, 0, stream>>>(xT, Wkb, bk, Kw, CCH, RCH, 0);
  proj_bf16_gemm<<<dim3(NTOK / 64, CCH / 128, BATCH), blk, 0, stream>>>(xT, Wvb, bv, Vw, CCH, CCH, 1);
  // fused flash attention + residual
  flash_attn<<<dim3(NTOK / 32, BATCH), blk, 0, stream>>>(Qw, Kw, Vw, x, gamma, out);
}
